// GCN_regressor_3092376453282
// MI455X (gfx1250) — matrix operand order repaired and checked by reading, not compiled, not run on hardware
//
#include <hip/hip_runtime.h>

#define NN      50000
#define NE      800000
#define H       128
#define NLAYERS 4
#define NGRAPHS 32
#define LN_EPS  1e-5f
#define MT      5                 // 16-row M-tiles per block => 80 rows/block, 50000/80 = 625 exact

typedef __attribute__((ext_vector_type(16))) __bf16 v16bf;
typedef __attribute__((ext_vector_type(8)))  __bf16 v8bf;
typedef __attribute__((ext_vector_type(4)))  __bf16 v4bf;
typedef __attribute__((ext_vector_type(8)))  float  v8f;
typedef __attribute__((vector_size(4 * sizeof(int)))) int v4i;   // matches builtin param type

#if defined(__AMDGCN__) && __has_builtin(__builtin_amdgcn_global_load_async_to_lds_b128)
#define HAVE_ASYNC_LDS 1
#else
#define HAVE_ASYNC_LDS 0
#endif

__device__ __forceinline__ void wait_async_zero() {
#if HAVE_ASYNC_LDS
#if __has_builtin(__builtin_amdgcn_s_wait_asynccnt)
  __builtin_amdgcn_s_wait_asynccnt(0);
#else
  asm volatile("s_wait_asynccnt 0" ::: "memory");
#endif
#endif
}

// ---------------- elementwise conversion f32 -> bf16 ----------------
__global__ void k_cvt_bf16(const float* __restrict__ src, __bf16* __restrict__ dst, int n) {
  int i = blockIdx.x * blockDim.x + threadIdx.x;
  if (i < n) dst[i] = (__bf16)src[i];
}

// ---------------- degree / normalization ----------------
__global__ void k_deg_init(float* __restrict__ deg) {
  int i = blockIdx.x * blockDim.x + threadIdx.x;
  if (i < NN) deg[i] = 1.0f;            // self-loop contribution
}
__global__ void k_deg_accum(const int* __restrict__ col, float* __restrict__ deg) {
  int e = blockIdx.x * blockDim.x + threadIdx.x;
  if (e < NE) unsafeAtomicAdd(&deg[col[e]], 1.0f);
}
__global__ void k_rsqrt_inplace(float* __restrict__ d) {
  int i = blockIdx.x * blockDim.x + threadIdx.x;
  if (i < NN) d[i] = rsqrtf(d[i]);      // deg >= 1 always (self-loops)
}

// ---------------- WMMA bf16 GEMM: C[M,128] = A[M,128] @ W[128,128]^T (+bias) ----------------
// Block tile: 80 rows (5 x 16 M-tiles) staged in LDS (async-to-LDS); 8 waves each own a
// 16-wide N tile; B fragments for all 4 K-steps preloaded once and reused across 5 M-tiles.
__global__ void __launch_bounds__(256)
k_gemm_bf16(const __bf16* __restrict__ A,   // [M,H] row-major
            const __bf16* __restrict__ W,   // [H,H] row-major, row n over K
            const float*  __restrict__ bias,// [H] or nullptr
            float* __restrict__ C,          // [M,H] f32 out
            __bf16* __restrict__ Cbf) {     // optional bf16 copy of C (may alias A; see note)
  __shared__ __bf16 Alds[16 * MT][H + 8];   // +8 pad: row stride 272B kills bank aliasing
  const int tid  = threadIdx.x;
  const int lane = tid & 31;
  const int wave = tid >> 5;
  const int m0   = blockIdx.x * (16 * MT);

  // stage 80x128 bf16 A tile: 256 threads x 5 x 16B chunks
#pragma unroll
  for (int j = 0; j < MT; ++j) {
    int ci = tid + j * 256;                 // 0..1279
    int r  = ci >> 4;                       // 0..79
    int cc = (ci & 15) * 8;                 // 0..120
    const __bf16* src = &A[(size_t)(m0 + r) * H + cc];
#if HAVE_ASYNC_LDS
    __builtin_amdgcn_global_load_async_to_lds_b128(
        (v4i*)src,
        (__attribute__((address_space(3))) v4i*)&Alds[r][cc],
        /*offset=*/0, /*cpol=*/0);
#else
    *(v8bf*)&Alds[r][cc] = *(const v8bf*)src;
#endif
  }
  wait_async_zero();
  __syncthreads();
  // NOTE: in-place Cbf==A is safe: this block's A rows are fully staged in LDS (barrier above)
  // before any store, and no other block touches these rows.

  const int n0 = wave * 16;
  const int ar = lane & 15;                 // A row within a 16-row M-tile
  const int ak = (lane >> 4) * 16;          // lanes 16-31 carry K+16..31
  const int bn = n0 + (lane & 15);          // B column (output channel)
  const int bk = (lane >> 4) * 16;          // lanes 16-31 carry K+16..31

  // preload B fragments for all 4 K-steps (reused across all 5 M-tiles)
  v16bf bfrag[4];
#pragma unroll
  for (int k4 = 0; k4 < 4; ++k4)
    bfrag[k4] = *(const v16bf*)&W[(size_t)bn * H + k4 * 32 + bk];

  v8f acc[MT];
#pragma unroll
  for (int mt = 0; mt < MT; ++mt) acc[mt] = {};

#pragma unroll
  for (int mt = 0; mt < MT; ++mt) {
#pragma unroll
    for (int k4 = 0; k4 < 4; ++k4) {
      const int kk = k4 * 32;
      v8bf alo = *(const v8bf*)&Alds[mt * 16 + ar][kk + ak];
      v8bf ahi = *(const v8bf*)&Alds[mt * 16 + ar][kk + ak + 8];
      v16bf a;
#pragma unroll
      for (int i = 0; i < 8; ++i) { a[i] = alo[i]; a[i + 8] = ahi[i]; }
      acc[mt] = __builtin_amdgcn_wmma_f32_16x16x32_bf16(
          /*neg_a=*/false, a, /*neg_b=*/false, bfrag[k4],
          /*c_mod=*/(short)0, acc[mt], /*reuse_a=*/false, /*reuse_b=*/false);
    }
  }

  const int cn   = n0 + (lane & 15);
  const float bv = bias ? bias[cn] : 0.0f;
#pragma unroll
  for (int mt = 0; mt < MT; ++mt) {
    const int mbase = m0 + mt * 16 + ((lane >> 4) << 3);
#pragma unroll
    for (int v = 0; v < 8; ++v) {
      float val = acc[mt][v] + bv;
      C[(size_t)(mbase + v) * H + cn] = val;
      if (Cbf) Cbf[(size_t)(mbase + v) * H + cn] = (__bf16)val;
    }
  }
}

// ---------------- agg init: bias + self-loop term ----------------
__global__ void k_agg_init(const float* __restrict__ t, const float* __restrict__ dis,
                           const float* __restrict__ bias, float* __restrict__ agg) {
  int i = blockIdx.x * blockDim.x + threadIdx.x;
  if (i < NN * H) {
    int n = i >> 7, c = i & (H - 1);
    float d = dis[n];
    agg[i] = bias[c] + t[i] * d * d;
  }
}

// ---------------- edge scatter: one wave per edge, float4 per lane, hw f32 atomics ----------------
__global__ void __launch_bounds__(256)
k_scatter(const int* __restrict__ row, const int* __restrict__ col,
          const float* __restrict__ dis, const float* __restrict__ t,
          float* __restrict__ agg) {
  int gid  = blockIdx.x * blockDim.x + threadIdx.x;
  int e    = gid >> 5;
  int lane = gid & 31;
  if (e >= NE) return;
  // prefetch a future edge's source row (global_prefetch_b8) to hide random-access latency
  int ep = e + 1024;
  if (ep < NE) __builtin_prefetch(&t[(size_t)row[ep] * H + lane * 4], 0, 1);
  int r = row[e], c = col[e];
  float nm = dis[r] * dis[c];
  const float4 tv = *(const float4*)&t[(size_t)r * H + lane * 4];
  float* dst = &agg[(size_t)c * H + lane * 4];
  unsafeAtomicAdd(dst + 0, tv.x * nm);
  unsafeAtomicAdd(dst + 1, tv.y * nm);
  unsafeAtomicAdd(dst + 2, tv.z * nm);
  unsafeAtomicAdd(dst + 3, tv.w * nm);
}

// ---------------- h = LayerNorm(h + relu(agg)), also emit bf16 copy : one wave per node ----------------
__global__ void __launch_bounds__(256)
k_res_relu_ln(float* __restrict__ h, const float* __restrict__ agg,
              const float* __restrict__ gamma, const float* __restrict__ beta,
              __bf16* __restrict__ hbf) {
  int node = blockIdx.x * 8 + (threadIdx.x >> 5);
  int lane = threadIdx.x & 31;
  size_t base = (size_t)node * H + lane * 4;
  float4 hv = *(const float4*)&h[base];
  float4 av = *(const float4*)&agg[base];
  float4 r;
  r.x = hv.x + fmaxf(av.x, 0.0f);
  r.y = hv.y + fmaxf(av.y, 0.0f);
  r.z = hv.z + fmaxf(av.z, 0.0f);
  r.w = hv.w + fmaxf(av.w, 0.0f);
  float s  = r.x + r.y + r.z + r.w;
  float s2 = r.x * r.x + r.y * r.y + r.z * r.z + r.w * r.w;
#pragma unroll
  for (int m = 16; m; m >>= 1) {
    s  += __shfl_xor(s,  m, 32);
    s2 += __shfl_xor(s2, m, 32);
  }
  float mean = s * (1.0f / H);
  float var  = s2 * (1.0f / H) - mean * mean;
  float inv  = rsqrtf(var + LN_EPS);
  float4 g = *(const float4*)&gamma[lane * 4];
  float4 b = *(const float4*)&beta[lane * 4];
  float4 y;
  y.x = (r.x - mean) * inv * g.x + b.x;
  y.y = (r.y - mean) * inv * g.y + b.y;
  y.z = (r.z - mean) * inv * g.z + b.z;
  y.w = (r.w - mean) * inv * g.w + b.w;
  *(float4*)&h[base] = y;
  v4bf yb;
  yb[0] = (__bf16)y.x; yb[1] = (__bf16)y.y; yb[2] = (__bf16)y.z; yb[3] = (__bf16)y.w;
  *(v4bf*)&hbf[base] = yb;
}

// ---------------- output projection + graph pooling ----------------
__global__ void k_zero_pool(float* __restrict__ pool) {
  int i = threadIdx.x;
  if (i < 2 * NGRAPHS) pool[i] = 0.0f;
}
__global__ void __launch_bounds__(256)
k_out_pool(const float* __restrict__ h, const float* __restrict__ Wout,
           const int* __restrict__ batch, float* __restrict__ pool) {
  int node = blockIdx.x * 8 + (threadIdx.x >> 5);
  int lane = threadIdx.x & 31;
  const float4 hv = *(const float4*)&h[(size_t)node * H + lane * 4];
  const float4 wv = *(const float4*)&Wout[lane * 4];
  float s = hv.x * wv.x + hv.y * wv.y + hv.z * wv.z + hv.w * wv.w;
#pragma unroll
  for (int m = 16; m; m >>= 1) s += __shfl_xor(s, m, 32);
  if (lane == 0) {
    int g = batch[node];
    unsafeAtomicAdd(&pool[g], s);
    unsafeAtomicAdd(&pool[NGRAPHS + g], 1.0f);
  }
}
__global__ void k_finalize(const float* __restrict__ pool, float* __restrict__ out) {
  int g = threadIdx.x;
  if (g < NGRAPHS) out[g] = pool[g] / fmaxf(pool[NGRAPHS + g], 1.0f);
}

// ---------------- host orchestration ----------------
extern "C" void kernel_launch(void* const* d_in, const int* in_sizes, int n_in,
                              void* d_out, int out_size, void* d_ws, size_t ws_size,
                              hipStream_t stream) {
  const float* x     = (const float*)d_in[0];
  const int*   ei    = (const int*)d_in[1];
  const int*   batch = (const int*)d_in[2];
  const float* W_emb = (const float*)d_in[3];
  const float* b_emb = (const float*)d_in[4];
  const float* W_l   = (const float*)d_in[5];
  const float* b_l   = (const float*)d_in[6];
  const float* gam   = (const float*)d_in[7];
  const float* bet   = (const float*)d_in[8];
  const float* W_out = (const float*)d_in[9];
  const int* row = ei;
  const int* col = ei + NE;

  // carve workspace (256B aligned regions)
  size_t off = 0;
  auto carve = [&](size_t bytes) -> char* {
    char* p = (char*)d_ws + off;
    off = (off + bytes + 255) & ~(size_t)255;
    return p;
  };
  __bf16* abf  = (__bf16*)carve((size_t)NN * H * sizeof(__bf16));          // bf16 activations
  __bf16* wbf  = (__bf16*)carve((size_t)(1 + NLAYERS) * H * H * sizeof(__bf16));
  float*  hbuf = (float*)carve((size_t)NN * H * sizeof(float));
  float*  tbuf = (float*)carve((size_t)NN * H * sizeof(float));
  float*  agg  = (float*)carve((size_t)NN * H * sizeof(float));
  float*  dis  = (float*)carve((size_t)NN * sizeof(float));
  float*  pool = (float*)carve(2 * NGRAPHS * sizeof(float));

  const int B = 256;
  const int gNH   = (NN * H + B - 1) / B;      // elementwise over node features
  const int gN    = (NN + B - 1) / B;
  const int gE    = (NE + B - 1) / B;
  const int gGemm = NN / (16 * MT);             // 625 (exact)
  const int gWave = NN / 8;                     // one wave per node, 8 waves/block (exact)
  const int gScat = (NE * 32 + B - 1) / B;      // one wave per edge

  // normalization coefficients
  k_deg_init<<<gN, B, 0, stream>>>(dis);
  k_deg_accum<<<gE, B, 0, stream>>>(col, dis);
  k_rsqrt_inplace<<<gN, B, 0, stream>>>(dis);

  // weights -> bf16
  k_cvt_bf16<<<(H * H + B - 1) / B, B, 0, stream>>>(W_emb, wbf, H * H);
  k_cvt_bf16<<<(NLAYERS * H * H + B - 1) / B, B, 0, stream>>>(W_l, wbf + H * H, NLAYERS * H * H);

  // embedding: h = x @ W_emb^T + b_emb ; also emit bf16 copy of h in-place into abf
  k_cvt_bf16<<<gNH, B, 0, stream>>>(x, abf, NN * H);
  k_gemm_bf16<<<gGemm, B, 0, stream>>>(abf, wbf, b_emb, hbuf, abf);

  for (int l = 0; l < NLAYERS; ++l) {
    k_gemm_bf16<<<gGemm, B, 0, stream>>>(abf, wbf + (size_t)(1 + l) * H * H, nullptr, tbuf, nullptr);
    k_agg_init<<<gNH, B, 0, stream>>>(tbuf, dis, b_l + l * H, agg);
    k_scatter<<<gScat, B, 0, stream>>>(row, col, dis, tbuf, agg);
    k_res_relu_ln<<<gWave, B, 0, stream>>>(hbuf, agg, gam + l * H, bet + l * H, abf);
  }

  k_zero_pool<<<1, 64, 0, stream>>>(pool);
  k_out_pool<<<gWave, B, 0, stream>>>(hbuf, W_out, batch, pool);
  k_finalize<<<1, NGRAPHS, 0, stream>>>(pool, (float*)d_out);
}
